// AFT_FULL_62440234549812
// MI455X (gfx1250) — matrix operand order repaired and checked by reading, not compiled, not run on hardware
//
#include <hip/hip_runtime.h>

// ---------------------------------------------------------------------------
// AFT-full for MI455X (gfx1250, wave32, WMMA).
//   q = sigmoid(x Wq^T + bq); k = x Wk^T + bk; v = x Wv^T + bv
//   P[t,j]  = exp(pb[t,j] - rowmax_j pb[t,j])          [N,N]  bf16
//   E[j,d]  = exp(k[b,j,d] - colmax_j k[b,j,d])        (stored transposed)
//   num = P @ (E*v), den = P @ E, out = q * num / den
// Matmuls: v_wmma_f32_16x16x32_bf16. Operands pre-converted to bf16 so the
// GEMM hot loops are pure load->wmma. Stage-2 stages P tiles into LDS via
// global_load_async_to_lds_b128 (double-buffered, s_wait_asynccnt).
// ---------------------------------------------------------------------------

#define BS  4
#define SEQ 512
#define DIM 128

typedef __attribute__((ext_vector_type(16))) __bf16 v16bf;
typedef __attribute__((ext_vector_type(8)))  __bf16 v8bf;
typedef __attribute__((ext_vector_type(4)))  __bf16 v4bf;
typedef __attribute__((ext_vector_type(8)))  float  v8f;

// Async-copy pointer types: builtin expects int4-vector pointers,
// global (AS1) source and LDS (AS3) destination.
typedef int v4i __attribute__((vector_size(16)));
typedef __attribute__((address_space(1))) v4i* gv4i_p;
typedef __attribute__((address_space(3))) v4i* lv4i_p;

#if defined(__has_builtin)
#if __has_builtin(__builtin_amdgcn_global_load_async_to_lds_b128) && \
    __has_builtin(__builtin_amdgcn_s_wait_asynccnt)
#define HAVE_ASYNC 1
#endif
#endif
#ifndef HAVE_ASYNC
#define HAVE_ASYNC 0
#endif

#define WMMA_BF16(a, b, c) \
  __builtin_amdgcn_wmma_f32_16x16x32_bf16(false, (a), false, (b), (short)0, (c), false, false)

// --- fragment loaders (bf16 memory) ----------------------------------------
// A fragment (16x32): per lane two contiguous 8-element runs along K.
// B fragment (32x16): per lane one contiguous 16-element run along K.
__device__ inline v16bf ld_2run8_bf(const __bf16* p, const __bf16* q) {
  v8bf lo = *(const v8bf*)p;
  v8bf hi = *(const v8bf*)q;
  v16bf r;
#pragma unroll
  for (int i = 0; i < 8; ++i) { r[i] = lo[i]; r[8 + i] = hi[i]; }
  return r;
}

__device__ inline v16bf ld_run16_bf(const __bf16* p) {
  v8bf lo = *(const v8bf*)p;
  v8bf hi = *(const v8bf*)(p + 8);
  v16bf r;
#pragma unroll
  for (int i = 0; i < 8; ++i) { r[i] = lo[i]; r[8 + i] = hi[i]; }
  return r;
}

// ---------------------------------------------------------------------------
// Kernel 0: one-shot f32 -> bf16 conversion (keeps cvt out of GEMM loops).
// ---------------------------------------------------------------------------
__global__ __launch_bounds__(256) void k_cvt(const float* __restrict__ in,
                                             __bf16* __restrict__ out, int n) {
  const int i = (blockIdx.x * 256 + threadIdx.x) * 4;
  if (i < n) {
    float4 f = *(const float4*)(in + i);
    v4bf o;
    o[0] = (__bf16)f.x; o[1] = (__bf16)f.y; o[2] = (__bf16)f.z; o[3] = (__bf16)f.w;
    *(v4bf*)(out + i) = o;
  }
}

// ---------------------------------------------------------------------------
// Kernel 1: projections. One wave per 16x16 output tile of [BS*SEQ, DIM].
// Pure bf16 load -> wmma; epilogue adds bias, applies sigmoid to q, and
// stores k,v TRANSPOSED f32 ([BS][DIM][SEQ]) for the stage-2 B operands.
// ---------------------------------------------------------------------------
__global__ __launch_bounds__(32) void k_proj(
    const __bf16* __restrict__ xbf,                    // [BS*SEQ, DIM]
    const __bf16* __restrict__ Wq, const float* __restrict__ bq,
    const __bf16* __restrict__ Wk, const float* __restrict__ bk,
    const __bf16* __restrict__ Wv, const float* __restrict__ bv,
    float* __restrict__ qS,                            // [BS*SEQ, DIM]
    float* __restrict__ kT,                            // [BS][DIM][SEQ]
    float* __restrict__ vT)                            // [BS][DIM][SEQ]
{
  const int et = blockIdx.x;           // e-tile: 0..7
  const int rt = blockIdx.y;           // row-tile: 0..127
  const int l  = threadIdx.x;
  const int g0 = l >> 4;
  const int ln = l & 15;
  const int e0 = et * 16, row0 = rt * 16;

  const int arow = row0 + ln;          // A row (M)
  const int bcol = e0 + ln;            // B col (N == output feature e)

  v8f accq = {}, acck = {}, accv = {};

#pragma unroll
  for (int ks = 0; ks < DIM; ks += 32) {
    const __bf16* ap = xbf + arow * DIM + ks + 16 * g0;
    v16bf a = ld_2run8_bf(ap, ap + 8);

    const int wof = bcol * DIM + ks + 16 * g0;   // W[e][d], row-major
    v16bf fq = ld_run16_bf(Wq + wof);
    v16bf fk = ld_run16_bf(Wk + wof);
    v16bf fv = ld_run16_bf(Wv + wof);

    accq = WMMA_BF16(a, fq, accq);
    acck = WMMA_BF16(a, fk, acck);
    accv = WMMA_BF16(a, fv, accv);
  }

  const float biq = bq[bcol], bik = bk[bcol], biv = bv[bcol];
  const int b = row0 >> 9;                   // whole tile lies in one batch
#pragma unroll
  for (int r = 0; r < 8; ++r) {
    const int grow = row0 + r + 8 * g0;      // C-layout: M = r + 8*(lane>=16)
    const int n = grow & (SEQ - 1);
    const float qv = accq[r] + biq;
    qS[grow * DIM + bcol] = 1.0f / (1.0f + __expf(-qv));
    kT[(b * DIM + bcol) * SEQ + n] = acck[r] + bik;
    vT[(b * DIM + bcol) * SEQ + n] = accv[r] + biv;
  }
}

// ---------------------------------------------------------------------------
// Kernel 2: per-row max + exp -> bf16. One 256-thread block per row of 512.
// ---------------------------------------------------------------------------
__global__ __launch_bounds__(256) void k_rowexp(
    const float* __restrict__ in,    // [R, 512]
    const float* __restrict__ mul,   // [R, 512] or null
    __bf16* __restrict__ out1,       // [R, 512]
    __bf16* __restrict__ out2,       // [R, 512] (iff mul)
    int hasMul)
{
  __shared__ float red[256];
  const int r = blockIdx.x;
  const int t = threadIdx.x;
  const float* row = in + r * SEQ;
  const float v0 = row[t], v1 = row[t + 256];

  red[t] = fmaxf(v0, v1);
  __syncthreads();
#pragma unroll
  for (int s = 128; s > 0; s >>= 1) {
    if (t < s) red[t] = fmaxf(red[t], red[t + s]);
    __syncthreads();
  }
  const float m = red[0];

  const float e0 = __expf(v0 - m), e1 = __expf(v1 - m);
  out1[r * SEQ + t]       = (__bf16)e0;
  out1[r * SEQ + t + 256] = (__bf16)e1;
  if (hasMul) {
    const float* mr = mul + r * SEQ;
    out2[r * SEQ + t]       = (__bf16)(e0 * mr[t]);
    out2[r * SEQ + t + 256] = (__bf16)(e1 * mr[t + 256]);
  }
}

// ---------------------------------------------------------------------------
// Kernel 3: num = P @ EV, den = P @ E (per batch), out = q*num/den.
// One wave per 32x32 (t,d) block: 2x2 register tiles, A-frags reused across
// both d-tiles -> 8 wmma per 32-deep k-step. P tile staged to LDS via async
// copy (double-buffered) when available.
// ---------------------------------------------------------------------------
__global__ __launch_bounds__(32) void k_aft(
    const __bf16* __restrict__ P,    // [SEQ, SEQ]
    const __bf16* __restrict__ ET,   // [BS][DIM][SEQ]
    const __bf16* __restrict__ EVT,  // [BS][DIM][SEQ]
    const float*  __restrict__ qS,   // [BS*SEQ, DIM]
    float* __restrict__ out)         // [BS, SEQ, DIM]
{
  const int db = blockIdx.x;   // 0..3  (32 d-cols)
  const int tb = blockIdx.y;   // 0..15 (32 t-rows)
  const int b  = blockIdx.z;   // 0..3
  const int l  = threadIdx.x;
  const int g0 = l >> 4;
  const int ln = l & 15;
  const int d0 = db * 32, t0 = tb * 32;

  const __bf16* E0  = ET  + (b * DIM + d0 + ln) * SEQ;
  const __bf16* E1  = ET  + (b * DIM + d0 + 16 + ln) * SEQ;
  const __bf16* EV0 = EVT + (b * DIM + d0 + ln) * SEQ;
  const __bf16* EV1 = EVT + (b * DIM + d0 + 16 + ln) * SEQ;

  v8f num00 = {}, num01 = {}, num10 = {}, num11 = {};
  v8f den00 = {}, den01 = {}, den10 = {}, den11 = {};

#if HAVE_ASYNC
  // LDS stage: 2 buffers of a 32x32 bf16 P tile (row stride 32 elem = 64B).
  __shared__ alignas(16) __bf16 stage[2][32 * 32];
  const __bf16* prow = P + (t0 + l) * SEQ;       // each lane owns one row
  {
    __bf16* dst = &stage[0][l * 32];
#pragma unroll
    for (int i = 0; i < 4; ++i)
      __builtin_amdgcn_global_load_async_to_lds_b128(
          (gv4i_p)(prow + i * 8), (lv4i_p)(dst + i * 8), 0, 0);
  }
#endif

  for (int js = 0; js < SEQ / 32; ++js) {
    const int j0 = js * 32;

#if HAVE_ASYNC
    if (js + 1 < SEQ / 32) {
      const __bf16* src = prow + j0 + 32;
      __bf16* dst = &stage[(js + 1) & 1][l * 32];
#pragma unroll
      for (int i = 0; i < 4; ++i)
        __builtin_amdgcn_global_load_async_to_lds_b128(
            (gv4i_p)(src + i * 8), (lv4i_p)(dst + i * 8), 0, 0);
      __builtin_amdgcn_s_wait_asynccnt(4);   // current tile's 4 copies done
    } else {
      __builtin_amdgcn_s_wait_asynccnt(0);
    }
    const __bf16* sA = &stage[js & 1][0];
    v16bf a0 = ld_2run8_bf(sA + ln * 32 + 16 * g0,
                           sA + ln * 32 + 8 + 16 * g0);
    v16bf a1 = ld_2run8_bf(sA + (16 + ln) * 32 + 16 * g0,
                           sA + (16 + ln) * 32 + 8 + 16 * g0);
#else
    const __bf16* pr0 = P + (t0 + ln) * SEQ + j0;
    const __bf16* pr1 = P + (t0 + 16 + ln) * SEQ + j0;
    v16bf a0 = ld_2run8_bf(pr0 + 16 * g0, pr0 + 8 + 16 * g0);
    v16bf a1 = ld_2run8_bf(pr1 + 16 * g0, pr1 + 8 + 16 * g0);
#endif

    if (j0 + 32 < SEQ) {                      // prefetch next B stream chunk
      __builtin_prefetch(E0 + j0 + 32 + 16 * g0, 0, 0);
      __builtin_prefetch(EV0 + j0 + 32 + 16 * g0, 0, 0);
    }

    v16bf fE0  = ld_run16_bf(E0  + j0 + 16 * g0);
    v16bf fE1  = ld_run16_bf(E1  + j0 + 16 * g0);
    v16bf fEV0 = ld_run16_bf(EV0 + j0 + 16 * g0);
    v16bf fEV1 = ld_run16_bf(EV1 + j0 + 16 * g0);

    den00 = WMMA_BF16(a0, fE0,  den00);
    den01 = WMMA_BF16(a0, fE1,  den01);
    den10 = WMMA_BF16(a1, fE0,  den10);
    den11 = WMMA_BF16(a1, fE1,  den11);
    num00 = WMMA_BF16(a0, fEV0, num00);
    num01 = WMMA_BF16(a0, fEV1, num01);
    num10 = WMMA_BF16(a1, fEV0, num10);
    num11 = WMMA_BF16(a1, fEV1, num11);
  }

#pragma unroll
  for (int tt = 0; tt < 2; ++tt) {
#pragma unroll
    for (int dd = 0; dd < 2; ++dd) {
      v8f nm = tt == 0 ? (dd == 0 ? num00 : num01) : (dd == 0 ? num10 : num11);
      v8f dn = tt == 0 ? (dd == 0 ? den00 : den01) : (dd == 0 ? den10 : den11);
      const int dcol = d0 + dd * 16 + ln;
#pragma unroll
      for (int r = 0; r < 8; ++r) {
        const int t = t0 + tt * 16 + r + 8 * g0;
        const float qv = qS[(b * SEQ + t) * DIM + dcol];
        out[(b * SEQ + t) * DIM + dcol] = qv * (nm[r] / dn[r]);
      }
    }
  }
}

// ---------------------------------------------------------------------------
extern "C" void kernel_launch(void* const* d_in, const int* in_sizes, int n_in,
                              void* d_out, int out_size, void* d_ws, size_t ws_size,
                              hipStream_t stream) {
  const float* x  = (const float*)d_in[0];
  const float* Wq = (const float*)d_in[1];
  const float* bq = (const float*)d_in[2];
  const float* Wk = (const float*)d_in[3];
  const float* bk = (const float*)d_in[4];
  const float* Wv = (const float*)d_in[5];
  const float* bv = (const float*)d_in[6];
  const float* pb = (const float*)d_in[7];
  float* out = (float*)d_out;

  char* ws = (char*)d_ws;
  const size_t MB = 1u << 20;
  const size_t KB = 1u << 10;
  float*  kT  = (float*) (ws + 0 * MB);               // 1 MB [BS][DIM][SEQ]
  float*  vT  = (float*) (ws + 1 * MB);               // 1 MB
  float*  qS  = (float*) (ws + 2 * MB);               // 1 MB [BS*SEQ][DIM]
  __bf16* P   = (__bf16*)(ws + 3 * MB);               // 512 KB [SEQ][SEQ]
  __bf16* ET  = (__bf16*)(ws + 3 * MB + 512 * KB);    // 512 KB
  __bf16* EVT = (__bf16*)(ws + 4 * MB);               // 512 KB
  __bf16* xbf = (__bf16*)(ws + 4 * MB + 512 * KB);    // 512 KB [BS*SEQ][DIM]
  __bf16* Wqb = (__bf16*)(ws + 5 * MB);               // 32 KB each
  __bf16* Wkb = (__bf16*)(ws + 5 * MB + 32 * KB);
  __bf16* Wvb = (__bf16*)(ws + 5 * MB + 64 * KB);

  // 0) one-shot bf16 conversions (x and the three weights)
  k_cvt<<<dim3((BS * SEQ * DIM / 4) / 256), 256, 0, stream>>>(x, xbf, BS * SEQ * DIM);
  k_cvt<<<dim3((DIM * DIM / 4 + 255) / 256), 256, 0, stream>>>(Wq, Wqb, DIM * DIM);
  k_cvt<<<dim3((DIM * DIM / 4 + 255) / 256), 256, 0, stream>>>(Wk, Wkb, DIM * DIM);
  k_cvt<<<dim3((DIM * DIM / 4 + 255) / 256), 256, 0, stream>>>(Wv, Wvb, DIM * DIM);

  // 1) projections (WMMA): q(sigmoid), k^T, v^T
  k_proj<<<dim3(DIM / 16, (BS * SEQ) / 16), 32, 0, stream>>>(
      xbf, Wqb, bq, Wkb, bk, Wvb, bv, qS, kT, vT);

  // 2a) E^T, (E*v)^T : rows are (b,d) pairs -> B*D = 512 rows
  k_rowexp<<<dim3(BS * DIM), 256, 0, stream>>>(kT, vT, ET, EVT, 1);
  // 2b) P : rows are t -> N = 512 rows
  k_rowexp<<<dim3(SEQ), 256, 0, stream>>>(pb, nullptr, P, nullptr, 0);

  // 3) num/den GEMMs (WMMA, 2x2 register tiles, async-staged A) + epilogue
  k_aft<<<dim3(DIM / 32, SEQ / 32, BS), 32, 0, stream>>>(P, ET, EVT, qS, out);
}
